// Glm4MoeDecoderLayer_64939905516055
// MI455X (gfx1250) — compile-verified, matrix operand order checked by reading, not run on hardware
//
#include <hip/hip_runtime.h>
#include <cstdint>

// ---------------- problem constants ----------------
constexpr int CB  = 2;       // batch
constexpr int CS  = 1024;    // seq
constexpr int CH  = 1024;    // hidden
constexpr int CNH = 8, CNKV = 2, CD = 128;
constexpr int CROT = 64;
constexpr int CE = 64, CK = 8, CNG = 8, CTG = 4;
constexpr int CFI = 512, CFSH = 512, CCAP = 512;
constexpr int CT  = CB * CS;                 // 2048 tokens
constexpr int CQO = (CNH + 2 * CNKV) * CD;   // 1536 qkv cols
constexpr float CEPS = 1e-5f;
constexpr float CSCALE = 0.08838834764831845f;   // D^-0.5
constexpr float LN_THETA = 9.210340371976184f;   // ln(10000)

#define NEG_INF (-__builtin_inff())

typedef __attribute__((ext_vector_type(16))) __bf16 bf16x16;
typedef __attribute__((ext_vector_type(8)))  __bf16 bf16x8;
typedef __attribute__((ext_vector_type(8)))  float  floatx8;

#define WMMA_BF16(a, b, c) \
  __builtin_amdgcn_wmma_f32_16x16x32_bf16(false, (a), false, (b), (short)0, (c), false, false)

__device__ inline floatx8 zero8() {
  floatx8 z = {0.f, 0.f, 0.f, 0.f, 0.f, 0.f, 0.f, 0.f};
  return z;
}

// 16-bit 16x32 A/B fragment from K-contiguous bf16: per lane two contiguous
// 8-element chunks (K = kb+0..7 and K = 16+kb+0..7) -> two b128 loads.
__device__ inline bf16x16 load_frag(const __bf16* __restrict__ p, int ld,
                                    int r0, int k0, int lane) {
  const __bf16* base = p + (size_t)(r0 + (lane & 15)) * ld + k0 + ((lane >> 4) << 3);
  bf16x8 lo = *(const bf16x8*)base;
  bf16x8 hi = *(const bf16x8*)(base + 16);
  return __builtin_shufflevector(lo, hi, 0, 1, 2, 3, 4, 5, 6, 7,
                                 8, 9, 10, 11, 12, 13, 14, 15);
}

template <int TM, int TN>
__device__ inline void mma_loop(const __bf16* __restrict__ A, int lda,
                                const __bf16* __restrict__ B, int ldb,
                                int r0, int c0, int K, int lane,
                                floatx8 (&acc)[TM][TN]) {
  for (int kk = 0; kk < K; kk += 32) {
    bf16x16 af[TM], bfr[TN];
#pragma unroll
    for (int i = 0; i < TM; ++i) af[i] = load_frag(A, lda, r0 + 16 * i, kk, lane);
#pragma unroll
    for (int j = 0; j < TN; ++j) bfr[j] = load_frag(B, ldb, c0 + 16 * j, kk, lane);
#pragma unroll
    for (int i = 0; i < TM; ++i)
#pragma unroll
      for (int j = 0; j < TN; ++j)
        acc[i][j] = WMMA_BF16(af[i], bfr[j], acc[i][j]);
  }
}

__device__ inline int wave_id() { return (int)((blockIdx.x * blockDim.x + threadIdx.x) >> 5); }
__device__ inline int lane_id() { return (int)(threadIdx.x & 31); }

// ---------------- weight preparation ----------------
// elementwise f32 -> bf16 (layout already [N,K], K-contiguous)
__global__ void convert_bf16_kernel(const float* __restrict__ src,
                                    __bf16* __restrict__ dst, int n) {
  for (int i = blockIdx.x * 256 + threadIdx.x; i < n; i += gridDim.x * 256)
    dst[i] = (__bf16)src[i];
}

// batched tiled transpose + convert: src[b][r][c] (f32, row stride src_rs)
// -> dst[b][c][r] (bf16, ld = rows)
__global__ void transpose_bf16_kernel(const float* __restrict__ src,
                                      __bf16* __restrict__ dst,
                                      int src_rs, int rows,
                                      size_t src_bs, size_t dst_bs) {
  __shared__ float tile[32][33];
  int b = blockIdx.z;
  int r0 = blockIdx.y * 32, c0 = blockIdx.x * 32;
  const float* s = src + (size_t)b * src_bs;
  __bf16* d = dst + (size_t)b * dst_bs;
#pragma unroll
  for (int i = 0; i < 4; ++i) {
    int r = threadIdx.y + i * 8;
    tile[r][threadIdx.x] = s[(size_t)(r0 + r) * src_rs + c0 + threadIdx.x];
  }
  __syncthreads();
#pragma unroll
  for (int i = 0; i < 4; ++i) {
    int c = threadIdx.y + i * 8;
    d[(size_t)(c0 + c) * rows + r0 + threadIdx.x] = (__bf16)tile[threadIdx.x][c];
  }
}

// ---------------- norm / rope ----------------
__global__ void rmsnorm_bf16_kernel(const float* __restrict__ x, const float* __restrict__ w,
                                    __bf16* __restrict__ out, int cols) {
  int row = blockIdx.x;
  const float* xr = x + (size_t)row * cols;
  __bf16* orow = out + (size_t)row * cols;
  __shared__ float red[256];
  float ss = 0.f;
  for (int c = threadIdx.x; c < cols; c += 256) { float v = xr[c]; ss += v * v; }
  red[threadIdx.x] = ss; __syncthreads();
  for (int st = 128; st > 0; st >>= 1) {
    if ((int)threadIdx.x < st) red[threadIdx.x] += red[threadIdx.x + st];
    __syncthreads();
  }
  float r = rsqrtf(red[0] / (float)cols + CEPS);
  for (int c = threadIdx.x; c < cols; c += 256) orow[c] = (__bf16)(xr[c] * r * w[c]);
}

// per-(token, head) RMSNorm + RoPE for q and k; writes bf16; 128 threads
__global__ void qk_rope_kernel(const float* __restrict__ qkv, const int* __restrict__ positions,
                               const float* __restrict__ qn_w, const float* __restrict__ kn_w,
                               __bf16* __restrict__ qnr, __bf16* __restrict__ knr) {
  int blk = blockIdx.x;
  int hs = blk % (CNH + CNKV);
  int t  = blk / (CNH + CNKV);
  int s  = t % CS;
  int d  = threadIdx.x;  // 0..127
  const float* src; const float* wgt; __bf16* dst;
  if (hs < CNH) {
    src = qkv + (size_t)t * CQO + hs * CD;
    wgt = qn_w;
    dst = qnr + ((size_t)t * CNH + hs) * CD;
  } else {
    int kv = hs - CNH;
    src = qkv + (size_t)t * CQO + CNH * CD + kv * CD;
    wgt = kn_w;
    dst = knr + ((size_t)t * CNKV + kv) * CD;
  }
  __shared__ float sh[CD];
  __shared__ float red[CD];
  float v = src[d];
  red[d] = v * v; __syncthreads();
  for (int st = 64; st > 0; st >>= 1) {
    if (d < st) red[d] += red[d + st];
    __syncthreads();
  }
  float r  = rsqrtf(red[0] / (float)CD + CEPS);
  float xn = v * r * wgt[d];
  sh[d] = xn; __syncthreads();
  float pos_f = (float)positions[s];
  float o;
  if (d < CROT / 2) {
    float freq = __expf(-((float)(2 * d) / (float)CROT) * LN_THETA);
    float ang = pos_f * freq;
    o = xn * __cosf(ang) - sh[d + CROT / 2] * __sinf(ang);
  } else if (d < CROT) {
    int i = d - CROT / 2;
    float freq = __expf(-((float)(2 * i) / (float)CROT) * LN_THETA);
    float ang = pos_f * freq;
    o = xn * __cosf(ang) + sh[d - CROT / 2] * __sinf(ang);
  } else {
    o = xn;
  }
  dst[d] = (__bf16)o;
}

// ---------------- GEMM kernels ----------------
// QKV projection (+bias): 32x32 per wave
__global__ void __launch_bounds__(256)
gemm_qkv_kernel(const __bf16* __restrict__ h, const __bf16* __restrict__ wqkvb,
                const float* __restrict__ bqkv, float* __restrict__ qkv) {
  int wid = wave_id(), lane = lane_id();
  constexpr int NTX = CQO / 32;  // 48
  int tx = wid % NTX, ty = wid / NTX;
  if (ty >= CT / 32) return;
  int r0 = ty * 32, c0 = tx * 32;
  floatx8 acc[2][2] = {{zero8(), zero8()}, {zero8(), zero8()}};
  mma_loop<2, 2>(h, CH, wqkvb, CH, r0, c0, CH, lane, acc);
  int n = lane & 15, mo = (lane >> 4) << 3;
#pragma unroll
  for (int i = 0; i < 2; ++i)
#pragma unroll
    for (int r = 0; r < 8; ++r) {
      int row = r0 + 16 * i + mo + r;
#pragma unroll
      for (int j = 0; j < 2; ++j) {
        int col = c0 + 16 * j + n;
        qkv[(size_t)row * CQO + col] = acc[i][j][r] + bqkv[col];
      }
    }
}

// attention scores (causal, scaled), f32 out; 16x16 per wave
__global__ void __launch_bounds__(256)
attn_scores_kernel(const __bf16* __restrict__ qnr, const __bf16* __restrict__ knr,
                   float* __restrict__ scores) {
  int wid = wave_id(), lane = lane_id();
  constexpr int NT = CS / 16;  // 64
  int txy = wid % (NT * NT);
  int bh  = wid / (NT * NT);
  if (bh >= CB * CNH) return;
  int tq = txy / NT, tk = txy % NT;
  int b = bh / CNH, hh = bh % CNH, kvh = hh / (CNH / CNKV);
  int q0 = tq * 16, k0c = tk * 16;
  int n = lane & 15, mo = (lane >> 4) << 3;
  float* out = scores + (size_t)bh * CS * CS;
  if (tk > tq) {  // fully masked tile: no compute
#pragma unroll
    for (int r = 0; r < 8; ++r)
      out[(size_t)(q0 + mo + r) * CS + k0c + n] = NEG_INF;
    return;
  }
  const __bf16* A  = qnr + (size_t)(b * CS) * (CNH * CD) + hh * CD;
  const __bf16* Bk = knr + (size_t)(b * CS) * (CNKV * CD) + kvh * CD;
  floatx8 acc[1][1] = {{zero8()}};
  mma_loop<1, 1>(A, CNH * CD, Bk, CNKV * CD, q0, k0c, CD, lane, acc);
#pragma unroll
  for (int r = 0; r < 8; ++r) {
    int sq = q0 + mo + r, sk = k0c + n;
    float v = acc[0][0][r] * CSCALE;
    out[(size_t)sq * CS + sk] = (sk <= sq) ? v : NEG_INF;
  }
}

// row softmax, emits bf16 probabilities
__global__ void softmax_probs_kernel(float* __restrict__ scores, __bf16* __restrict__ probs) {
  int row = blockIdx.x;  // CB*CNH*CS rows
  float* p = scores + (size_t)row * CS;
  __bf16* q = probs + (size_t)row * CS;
  __shared__ float red[256];
  float m = NEG_INF;
  for (int c = threadIdx.x; c < CS; c += 256) m = fmaxf(m, p[c]);
  red[threadIdx.x] = m; __syncthreads();
  for (int st = 128; st > 0; st >>= 1) {
    if ((int)threadIdx.x < st) red[threadIdx.x] = fmaxf(red[threadIdx.x], red[threadIdx.x + st]);
    __syncthreads();
  }
  m = red[0]; __syncthreads();
  float ssum = 0.f;
  for (int c = threadIdx.x; c < CS; c += 256) { float e = __expf(p[c] - m); p[c] = e; ssum += e; }
  red[threadIdx.x] = ssum; __syncthreads();
  for (int st = 128; st > 0; st >>= 1) {
    if ((int)threadIdx.x < st) red[threadIdx.x] += red[threadIdx.x + st];
    __syncthreads();
  }
  float inv = 1.f / red[0];
  for (int c = threadIdx.x; c < CS; c += 256) q[c] = (__bf16)(p[c] * inv);
}

// attn @ V: probs [S,S] bf16 x vt [D,S] bf16, causal K-truncation; 32x32/wave
__global__ void __launch_bounds__(256)
attn_out_kernel(const __bf16* __restrict__ probs, const __bf16* __restrict__ vt,
                __bf16* __restrict__ attnout) {
  int wid = wave_id(), lane = lane_id();
  constexpr int NTY = CS / 32, NTX = CD / 32;  // 32 x 4
  int tt = wid % (NTY * NTX);
  int bh = wid / (NTY * NTX);
  if (bh >= CB * CNH) return;
  int ty = tt / NTX, tx = tt % NTX;
  int b = bh / CNH, hh = bh % CNH, kvh = hh / (CNH / CNKV);
  const __bf16* A  = probs + (size_t)bh * CS * CS;
  const __bf16* Bv = vt + (size_t)(b * CNKV + kvh) * CD * CS;
  int r0 = ty * 32, c0 = tx * 32;
  floatx8 acc[2][2] = {{zero8(), zero8()}, {zero8(), zero8()}};
  mma_loop<2, 2>(A, CS, Bv, CS, r0, c0, r0 + 32, lane, acc);  // probs==0 past r0+31
  int n = lane & 15, mo = (lane >> 4) << 3;
#pragma unroll
  for (int i = 0; i < 2; ++i)
#pragma unroll
    for (int r = 0; r < 8; ++r) {
      int row = b * CS + r0 + 16 * i + mo + r;
#pragma unroll
      for (int j = 0; j < 2; ++j)
        attnout[(size_t)row * (CNH * CD) + hh * CD + c0 + 16 * j + n] =
            (__bf16)acc[i][j][r];
    }
}

// output projection + residual: 32x32/wave
__global__ void __launch_bounds__(256)
gemm_wo_kernel(const __bf16* __restrict__ attnout, const __bf16* __restrict__ wob,
               const float* __restrict__ x, float* __restrict__ h2) {
  int wid = wave_id(), lane = lane_id();
  constexpr int NTX = CH / 32;  // 32
  int tx = wid % NTX, ty = wid / NTX;
  if (ty >= CT / 32) return;
  int r0 = ty * 32, c0 = tx * 32;
  floatx8 acc[2][2] = {{zero8(), zero8()}, {zero8(), zero8()}};
  mma_loop<2, 2>(attnout, CNH * CD, wob, CNH * CD, r0, c0, CNH * CD, lane, acc);
  int n = lane & 15, mo = (lane >> 4) << 3;
#pragma unroll
  for (int i = 0; i < 2; ++i)
#pragma unroll
    for (int r = 0; r < 8; ++r) {
      int row = r0 + 16 * i + mo + r;
#pragma unroll
      for (int j = 0; j < 2; ++j) {
        size_t idx = (size_t)row * CH + c0 + 16 * j + n;
        h2[idx] = acc[i][j][r] + x[idx];
      }
    }
}

// router logits: 32x32/wave over [T, E]
__global__ void __launch_bounds__(256)
gemm_gate_kernel(const __bf16* __restrict__ xt, const __bf16* __restrict__ gwb,
                 float* __restrict__ logits) {
  int wid = wave_id(), lane = lane_id();
  constexpr int NTX = CE / 32;  // 2
  int tx = wid % NTX, ty = wid / NTX;
  if (ty >= CT / 32) return;
  int r0 = ty * 32, c0 = tx * 32;
  floatx8 acc[2][2] = {{zero8(), zero8()}, {zero8(), zero8()}};
  mma_loop<2, 2>(xt, CH, gwb, CH, r0, c0, CH, lane, acc);
  int n = lane & 15, mo = (lane >> 4) << 3;
#pragma unroll
  for (int i = 0; i < 2; ++i)
#pragma unroll
    for (int r = 0; r < 8; ++r) {
      int row = r0 + 16 * i + mo + r;
#pragma unroll
      for (int j = 0; j < 2; ++j)
        logits[(size_t)row * CE + c0 + 16 * j + n] = acc[i][j][r];
    }
}

// ---------------- routing ----------------
__global__ void route_kernel(const float* __restrict__ logits, const float* __restrict__ gate_b,
                             int* __restrict__ topi, float* __restrict__ fw) {
  int t = blockIdx.x * blockDim.x + threadIdx.x;
  if (t >= CT) return;
  float sc[CE], corr[CE];
  const float* lg = logits + (size_t)t * CE;
  for (int e = 0; e < CE; ++e) {
    float s = 1.f / (1.f + __expf(-lg[e]));
    sc[e] = s;
    corr[e] = s + gate_b[e];
  }
  float grp[CNG];
  for (int g = 0; g < CNG; ++g) {
    float m1 = NEG_INF, m2 = NEG_INF;
    for (int j = 0; j < CE / CNG; ++j) {
      float v = corr[g * (CE / CNG) + j];
      if (v > m1) { m2 = m1; m1 = v; } else if (v > m2) { m2 = v; }
    }
    grp[g] = m1 + m2;
  }
  bool gsel[CNG];
  for (int g = 0; g < CNG; ++g) gsel[g] = false;
  for (int it = 0; it < CTG; ++it) {
    int best = 0; float bv = NEG_INF;
    for (int g = 0; g < CNG; ++g)
      if (!gsel[g] && grp[g] > bv) { bv = grp[g]; best = g; }
    gsel[best] = true;
  }
  bool esel[CE];
  for (int e = 0; e < CE; ++e) esel[e] = false;
  float wsum = 0.f;
  for (int it = 0; it < CK; ++it) {
    int best = 0; float bv = NEG_INF;
    for (int e = 0; e < CE; ++e) {
      if (esel[e] || !gsel[e / (CE / CNG)]) continue;
      if (corr[e] > bv) { bv = corr[e]; best = e; }
    }
    esel[best] = true;
    topi[(size_t)t * CK + it] = best;
    float wv = sc[best];
    fw[(size_t)t * CK + it] = wv;
    wsum += wv;
  }
  float inv = 1.f / wsum;  // RSF = 1.0
  for (int it = 0; it < CK; ++it) fw[(size_t)t * CK + it] *= inv;
}

// deterministic capacity scatter (matches reference cumsum semantics)
__global__ void scatter_kernel(const int* __restrict__ topi, const float* __restrict__ fw,
                               int* __restrict__ slot_tid, float* __restrict__ slot_w,
                               int* __restrict__ entry_pos, int* __restrict__ counts) {
  int e = threadIdx.x;  // one block, 64 threads
  int cnt = 0;
  for (int i = 0; i < CT * CK; ++i) {
    if (topi[i] == e) {
      if (cnt < CCAP) {
        slot_tid[e * CCAP + cnt] = i / CK;
        slot_w[e * CCAP + cnt]   = fw[i];
        entry_pos[i] = cnt;
      } else {
        entry_pos[i] = -1;
      }
      cnt++;
    }
  }
  counts[e] = cnt < CCAP ? cnt : CCAP;
}

__global__ void buf_fill_kernel(const __bf16* __restrict__ xt, const int* __restrict__ slot_tid,
                                const int* __restrict__ counts, __bf16* __restrict__ buf) {
  int slot = blockIdx.x;
  int e = slot / CCAP, p = slot % CCAP;
  if (p >= counts[e]) return;
  int t = slot_tid[slot];
  const uint4* s4 = (const uint4*)(xt + (size_t)t * CH);
  uint4* d4 = (uint4*)(buf + (size_t)slot * CH);
  d4[threadIdx.x] = s4[threadIdx.x];  // 128 threads x 16B = 2KB row
}

// ---------------- expert GEMMs ----------------
// fused gate/up GEMM + SiLU*up; 32 rows x 16 cols per wave, dual accumulators
__global__ void __launch_bounds__(256)
moe_gu_kernel(const __bf16* __restrict__ buf, const __bf16* __restrict__ wgu_t,
              const int* __restrict__ counts, __bf16* __restrict__ act) {
  int wid = wave_id(), lane = lane_id();
  constexpr int NTY = CCAP / 32, NTX = CFI / 16;  // 16 x 32
  int e = wid / (NTY * NTX);
  if (e >= CE) return;
  int rem = wid % (NTY * NTX);
  int ty = rem / NTX, tx = rem % NTX;
  int r0 = ty * 32;
  if (r0 >= counts[e]) return;  // wave-uniform skip keeps EXEC all-1s
  int c0 = tx * 16;
  const __bf16* A  = buf + (size_t)e * CCAP * CH;
  const __bf16* Bt = wgu_t + (size_t)e * (2 * CFI) * CH;  // [2FI][H] bf16
  floatx8 ag[2] = {zero8(), zero8()};
  floatx8 au[2] = {zero8(), zero8()};
  for (int kk = 0; kk < CH; kk += 32) {
    if (kk + 32 < CH)
      __builtin_prefetch(Bt + (size_t)c0 * CH + kk + 32, 0, 1);
    bf16x16 a0 = load_frag(A, CH, r0, kk, lane);
    bf16x16 a1 = load_frag(A, CH, r0 + 16, kk, lane);
    bf16x16 bg = load_frag(Bt, CH, c0, kk, lane);
    bf16x16 bu = load_frag(Bt, CH, CFI + c0, kk, lane);
    ag[0] = WMMA_BF16(a0, bg, ag[0]);
    ag[1] = WMMA_BF16(a1, bg, ag[1]);
    au[0] = WMMA_BF16(a0, bu, au[0]);
    au[1] = WMMA_BF16(a1, bu, au[1]);
  }
  int n = lane & 15, mo = (lane >> 4) << 3;
  __bf16* out = act + (size_t)e * CCAP * CFI;
#pragma unroll
  for (int i = 0; i < 2; ++i)
#pragma unroll
    for (int r = 0; r < 8; ++r) {
      float g = ag[i][r], u = au[i][r];
      float silu = g / (1.f + __expf(-g));
      out[(size_t)(r0 + 16 * i + mo + r) * CFI + c0 + n] = (__bf16)(silu * u);
    }
}

// expert down GEMM: 32x32/wave -> f32 eo
__global__ void __launch_bounds__(256)
moe_dn_kernel(const __bf16* __restrict__ act, const __bf16* __restrict__ wdn_t,
              const int* __restrict__ counts, float* __restrict__ eo) {
  int wid = wave_id(), lane = lane_id();
  constexpr int NTY = CCAP / 32, NTX = CH / 32;  // 16 x 32
  int e = wid / (NTY * NTX);
  if (e >= CE) return;
  int rem = wid % (NTY * NTX);
  int ty = rem / NTX, tx = rem % NTX;
  int r0 = ty * 32;
  if (r0 >= counts[e]) return;
  int c0 = tx * 32;
  const __bf16* A  = act + (size_t)e * CCAP * CFI;
  const __bf16* Bt = wdn_t + (size_t)e * CH * CFI;  // [H][FI] bf16
  floatx8 acc[2][2] = {{zero8(), zero8()}, {zero8(), zero8()}};
  mma_loop<2, 2>(A, CFI, Bt, CFI, r0, c0, CFI, lane, acc);
  int n = lane & 15, mo = (lane >> 4) << 3;
  float* out = eo + (size_t)e * CCAP * CH;
#pragma unroll
  for (int i = 0; i < 2; ++i)
#pragma unroll
    for (int r = 0; r < 8; ++r) {
      int row = r0 + 16 * i + mo + r;
#pragma unroll
      for (int j = 0; j < 2; ++j)
        out[(size_t)row * CH + c0 + 16 * j + n] = acc[i][j][r];
    }
}

// ---------------- shared expert ----------------
__global__ void __launch_bounds__(256)
shared_gu_kernel(const __bf16* __restrict__ xt, const __bf16* __restrict__ swgu_t,
                 __bf16* __restrict__ actsh) {
  int wid = wave_id(), lane = lane_id();
  constexpr int NTX = CFSH / 16;  // 32
  int tx = wid % NTX, ty = wid / NTX;
  if (ty >= CT / 32) return;
  int r0 = ty * 32, c0 = tx * 16;
  floatx8 ag[2] = {zero8(), zero8()};
  floatx8 au[2] = {zero8(), zero8()};
  for (int kk = 0; kk < CH; kk += 32) {
    bf16x16 a0 = load_frag(xt, CH, r0, kk, lane);
    bf16x16 a1 = load_frag(xt, CH, r0 + 16, kk, lane);
    bf16x16 bg = load_frag(swgu_t, CH, c0, kk, lane);
    bf16x16 bu = load_frag(swgu_t, CH, CFSH + c0, kk, lane);
    ag[0] = WMMA_BF16(a0, bg, ag[0]);
    ag[1] = WMMA_BF16(a1, bg, ag[1]);
    au[0] = WMMA_BF16(a0, bu, au[0]);
    au[1] = WMMA_BF16(a1, bu, au[1]);
  }
  int n = lane & 15, mo = (lane >> 4) << 3;
#pragma unroll
  for (int i = 0; i < 2; ++i)
#pragma unroll
    for (int r = 0; r < 8; ++r) {
      float g = ag[i][r], u = au[i][r];
      float silu = g / (1.f + __expf(-g));
      actsh[(size_t)(r0 + 16 * i + mo + r) * CFSH + c0 + n] = (__bf16)(silu * u);
    }
}

__global__ void __launch_bounds__(256)
shared_dn_kernel(const __bf16* __restrict__ actsh, const __bf16* __restrict__ swdn_t,
                 float* __restrict__ shout) {
  int wid = wave_id(), lane = lane_id();
  constexpr int NTX = CH / 32;  // 32
  int tx = wid % NTX, ty = wid / NTX;
  if (ty >= CT / 32) return;
  int r0 = ty * 32, c0 = tx * 32;
  floatx8 acc[2][2] = {{zero8(), zero8()}, {zero8(), zero8()}};
  mma_loop<2, 2>(actsh, CFSH, swdn_t, CFSH, r0, c0, CFSH, lane, acc);
  int n = lane & 15, mo = (lane >> 4) << 3;
#pragma unroll
  for (int i = 0; i < 2; ++i)
#pragma unroll
    for (int r = 0; r < 8; ++r) {
      int row = r0 + 16 * i + mo + r;
#pragma unroll
      for (int j = 0; j < 2; ++j)
        shout[(size_t)row * CH + c0 + 16 * j + n] = acc[i][j][r];
    }
}

// ---------------- final: out = h2 + shared + sum_k w*eo ----------------
__global__ void final_kernel(const float* __restrict__ h2, const float* __restrict__ shout,
                             const float* __restrict__ eo, const int* __restrict__ topi,
                             const float* __restrict__ fw, const int* __restrict__ entry_pos,
                             float* __restrict__ out) {
  int t = blockIdx.x;
  const int*   ti  = topi + (size_t)t * CK;
  const float* fwt = fw + (size_t)t * CK;
  const int*   ep  = entry_pos + (size_t)t * CK;
  for (int c = threadIdx.x; c < CH; c += 256) {
    size_t idx = (size_t)t * CH + c;
    float v = h2[idx] + shout[idx];
#pragma unroll
    for (int j = 0; j < CK; ++j) {
      int p = ep[j];
      if (p >= 0) v += fwt[j] * eo[((size_t)ti[j] * CCAP + p) * CH + c];
    }
    out[idx] = v;
  }
}

// ---------------- host ----------------
extern "C" void kernel_launch(void* const* d_in, const int* in_sizes, int n_in,
                              void* d_out, int out_size, void* d_ws, size_t ws_size,
                              hipStream_t stream) {
  (void)in_sizes; (void)n_in; (void)out_size; (void)ws_size;
  const float* x         = (const float*)d_in[0];
  const int*   positions = (const int*)d_in[1];
  const float* ln1_w     = (const float*)d_in[2];
  const float* ln2_w     = (const float*)d_in[3];
  const float* wqkv      = (const float*)d_in[4];
  const float* bqkv      = (const float*)d_in[5];
  const float* qn_w      = (const float*)d_in[6];
  const float* kn_w      = (const float*)d_in[7];
  const float* wo        = (const float*)d_in[8];
  const float* gate_w    = (const float*)d_in[9];
  const float* gate_b    = (const float*)d_in[10];
  const float* w_gu      = (const float*)d_in[11];
  const float* w_dn      = (const float*)d_in[12];
  const float* sw_gu     = (const float*)d_in[13];
  const float* sw_dn     = (const float*)d_in[14];
  float* out = (float*)d_out;

  char* base = (char*)d_ws;
  size_t off = 0;
  auto alloc = [&](size_t bytes) -> char* {
    char* p = base + off;
    off += (bytes + 255) & ~(size_t)255;
    return p;
  };
  // bf16 weights (converted/transposed once per call)
  __bf16* wqkvb  = (__bf16*)alloc((size_t)CQO * CH * 2);
  __bf16* wob    = (__bf16*)alloc((size_t)CH * (CNH * CD) * 2);
  __bf16* gwb    = (__bf16*)alloc((size_t)CE * CH * 2);
  __bf16* wgu_t  = (__bf16*)alloc((size_t)CE * (2 * CFI) * CH * 2);
  __bf16* wdn_t  = (__bf16*)alloc((size_t)CE * CH * CFI * 2);
  __bf16* swgu_t = (__bf16*)alloc((size_t)(2 * CFSH) * CH * 2);
  __bf16* swdn_t = (__bf16*)alloc((size_t)CH * CFSH * 2);
  // activations
  __bf16* hbuf   = (__bf16*)alloc((size_t)CT * CH * 2);
  float*  qkv    = (float*)alloc((size_t)CT * CQO * 4);
  __bf16* qnr    = (__bf16*)alloc((size_t)CT * CNH * CD * 2);
  __bf16* knr    = (__bf16*)alloc((size_t)CT * CNKV * CD * 2);
  __bf16* vt     = (__bf16*)alloc((size_t)CB * CNKV * CD * CS * 2);
  float*  scores = (float*)alloc((size_t)CB * CNH * CS * CS * 4);
  __bf16* probs  = (__bf16*)alloc((size_t)CB * CNH * CS * CS * 2);
  __bf16* attno  = (__bf16*)alloc((size_t)CT * CNH * CD * 2);
  float*  h2     = (float*)alloc((size_t)CT * CH * 4);
  __bf16* xt     = (__bf16*)alloc((size_t)CT * CH * 2);
  float*  logits = (float*)alloc((size_t)CT * CE * 4);
  int*    topi   = (int*)alloc((size_t)CT * CK * 4);
  float*  fw     = (float*)alloc((size_t)CT * CK * 4);
  int*    entry_pos = (int*)alloc((size_t)CT * CK * 4);
  int*    counts = (int*)alloc((size_t)CE * 4);
  int*    slot_tid = (int*)alloc((size_t)CE * CCAP * 4);
  float*  slot_w = (float*)alloc((size_t)CE * CCAP * 4);
  __bf16* bufe   = (__bf16*)alloc((size_t)CE * CCAP * CH * 2);
  __bf16* acte   = (__bf16*)alloc((size_t)CE * CCAP * CFI * 2);
  float*  eo     = (float*)alloc((size_t)CE * CCAP * CH * 4);
  __bf16* actsh  = (__bf16*)alloc((size_t)CT * CFSH * 2);
  float*  shout  = (float*)alloc((size_t)CT * CH * 4);

  dim3 tb(32, 8);

  // ---- weight prep (independent of activations) ----
  convert_bf16_kernel<<<(CQO * CH) / 256, 256, 0, stream>>>(wqkv, wqkvb, CQO * CH);
  convert_bf16_kernel<<<(CH * CNH * CD) / 256, 256, 0, stream>>>(wo, wob, CH * CNH * CD);
  convert_bf16_kernel<<<(CE * CH) / 256, 256, 0, stream>>>(gate_w, gwb, CE * CH);
  transpose_bf16_kernel<<<dim3((2 * CFI) / 32, CH / 32, CE), tb, 0, stream>>>(
      w_gu, wgu_t, 2 * CFI, CH, (size_t)CH * 2 * CFI, (size_t)(2 * CFI) * CH);
  transpose_bf16_kernel<<<dim3(CH / 32, CFI / 32, CE), tb, 0, stream>>>(
      w_dn, wdn_t, CH, CFI, (size_t)CFI * CH, (size_t)CH * CFI);
  transpose_bf16_kernel<<<dim3((2 * CFSH) / 32, CH / 32, 1), tb, 0, stream>>>(
      sw_gu, swgu_t, 2 * CFSH, CH, 0, 0);
  transpose_bf16_kernel<<<dim3(CH / 32, CFSH / 32, 1), tb, 0, stream>>>(
      sw_dn, swdn_t, CH, CFSH, 0, 0);

  // ---- attention ----
  rmsnorm_bf16_kernel<<<CT, 256, 0, stream>>>(x, ln1_w, hbuf, CH);
  gemm_qkv_kernel<<<((CT / 32) * (CQO / 32)) / 8, 256, 0, stream>>>(hbuf, wqkvb, bqkv, qkv);
  // V slice -> [D,S] bf16 per (b,kv)
  for (int b = 0; b < CB; ++b)
    for (int kv = 0; kv < CNKV; ++kv)
      transpose_bf16_kernel<<<dim3(CD / 32, CS / 32, 1), tb, 0, stream>>>(
          qkv + (size_t)(b * CS) * CQO + (CNH + CNKV) * CD + kv * CD,
          vt + (size_t)(b * CNKV + kv) * CD * CS, CQO, CS, 0, 0);
  qk_rope_kernel<<<CT * (CNH + CNKV), CD, 0, stream>>>(qkv, positions, qn_w, kn_w, qnr, knr);
  attn_scores_kernel<<<(CB * CNH * (CS / 16) * (CS / 16)) / 8, 256, 0, stream>>>(qnr, knr, scores);
  softmax_probs_kernel<<<CB * CNH * CS, 256, 0, stream>>>(scores, probs);
  attn_out_kernel<<<(CB * CNH * (CS / 32) * (CD / 32)) / 8, 256, 0, stream>>>(probs, vt, attno);
  gemm_wo_kernel<<<((CT / 32) * (CH / 32)) / 8, 256, 0, stream>>>(attno, wob, x, h2);

  // ---- MoE ----
  rmsnorm_bf16_kernel<<<CT, 256, 0, stream>>>(h2, ln2_w, xt, CH);
  gemm_gate_kernel<<<((CT / 32) * (CE / 32)) / 8, 256, 0, stream>>>(xt, gwb, logits);
  route_kernel<<<CT / 64, 64, 0, stream>>>(logits, gate_b, topi, fw);
  scatter_kernel<<<1, CE, 0, stream>>>(topi, fw, slot_tid, slot_w, entry_pos, counts);
  buf_fill_kernel<<<CE * CCAP, 128, 0, stream>>>(xt, slot_tid, counts, bufe);
  moe_gu_kernel<<<(CE * (CCAP / 32) * (CFI / 16)) / 8, 256, 0, stream>>>(bufe, wgu_t, counts, acte);
  moe_dn_kernel<<<(CE * (CCAP / 32) * (CH / 32)) / 8, 256, 0, stream>>>(acte, wdn_t, counts, eo);
  shared_gu_kernel<<<((CT / 32) * (CFSH / 16)) / 8, 256, 0, stream>>>(xt, swgu_t, actsh);
  shared_dn_kernel<<<((CT / 32) * (CH / 32)) / 8, 256, 0, stream>>>(actsh, swdn_t, shout);
  final_kernel<<<CT, 256, 0, stream>>>(h2, shout, eo, topi, fw, entry_pos, out);
}
